// NRIModel_83013127897106
// MI455X (gfx1250) — compile-verified, matrix operand order checked by reading, not run on hardware
//
#include <hip/hip_runtime.h>
#include <hip/hip_bf16.h>
#include <cstdint>

// ---------------------------------------------------------------------------
// NRI forward pass for MI455X (gfx1250, wave32, WMMA).
// All GEMMs: v_wmma_f32_16x16x32_bf16, bf16 activations end-to-end, fp32
// accumulate, bias+ELU fused. Weights pre-packed per launch into the wave32
// B-fragment layout so the hot loop does only contiguous 32B loads.
// K is a template parameter -> fully unrolled straight-line K loop.
// ---------------------------------------------------------------------------

typedef float  v8f   __attribute__((ext_vector_type(8)));
typedef __bf16 v16bf __attribute__((ext_vector_type(16)));
typedef __bf16 v8bf  __attribute__((ext_vector_type(8)));

namespace {
constexpr int kNodes   = 16384;
constexpr int kEdges   = 131072;
constexpr int kD       = 128;
constexpr int kInFeat  = 16;
constexpr int kClasses = 4;
constexpr int kIters   = 10;
// gamma * lambda_amp = (1/(2*(1-0.1))) * 0.1
constexpr float kLam = 0.1f / (2.0f * (1.0f - 0.1f));
}

// ---------------------- weight packing ------------------------------------
// Pack W[K,N] (row-major fp32) into WMMA B-fragments (bf16), K padded to 32.
// Fragment f = kt*(N/16)+nt holds 32 lanes x 16 bf16; lane l, elem i maps to
//   k = kt*32 + (l>>4)*8 + (i>>3)*16 + (i&7),  n = nt*16 + (l&15)
// matching the 16-bit B-matrix (K x 16) wave32 layout (ISA 7.12.2).
__global__ void pack_weight(const float* __restrict__ W, __bf16* __restrict__ P,
                            int K, int N) {
  const int frag = blockIdx.x;
  const int lane = threadIdx.x;          // 0..31
  const int nTiles = N >> 4;
  const int kt = frag / nTiles;
  const int nt = frag - kt * nTiles;
  const int half = lane >> 4;
  const int col  = nt * 16 + (lane & 15);
  __bf16* o = P + ((size_t)frag * 32 + lane) * 16;
#pragma unroll
  for (int i = 0; i < 16; ++i) {
    const int k = kt * 32 + half * 8 + ((i >> 3) << 4) + (i & 7);
    const float v = (k < K) ? W[(size_t)k * N + col] : 0.0f;
    o[i] = (__bf16)v;
  }
}

// ---------------------- WMMA GEMM + bias + ELU -----------------------------
// AMODE 0: A = A0[M,KVAL] row-major bf16
// AMODE 1: A row r = concat(A0[sidx[r]], A0[didx[r]])            (KVAL = 256)
// AMODE 2: A row r = concat(A0[sidx[r]], A0[didx[r]], A2[r])     (KVAL = 384)
template <int AMODE, int KVAL>
__device__ __forceinline__ const __bf16* a_run_base(
    const __bf16* __restrict__ A0, const __bf16* __restrict__ A2,
    int s0, int s1, int row, int k0) {
  if constexpr (AMODE == 0) {
    return A0 + (size_t)row * KVAL + k0;
  } else if constexpr (AMODE == 1) {
    return (k0 < kD) ? (A0 + (size_t)s0 * kD + k0)
                     : (A0 + (size_t)s1 * kD + (k0 - kD));
  } else {
    if (k0 < kD)     return A0 + (size_t)s0 * kD + k0;
    if (k0 < 2 * kD) return A0 + (size_t)s1 * kD + (k0 - kD);
    return A2 + (size_t)row * kD + (k0 - 2 * kD);
  }
}

__device__ __forceinline__ void store_tile_elu(const v8f& acc, int mt, int nt,
                                               int lane, const float* __restrict__ bias,
                                               float* __restrict__ Outf,
                                               __bf16* __restrict__ Outb) {
  union { v8f v; float f[8]; } cu;
  cu.v = acc;
  const int col  = nt * 16 + (lane & 15);
  const int half = lane >> 4;
  const float bv = bias[col];
  float r[8];
#pragma unroll
  for (int i = 0; i < 8; ++i) {
    // C/D layout: VGPR i -> M = i (lanes 0-15) / i+8 (lanes 16-31), N = lane%16
    float v = cu.f[i] + bv;
    r[i] = (v > 0.0f) ? v : (__expf(v) - 1.0f);    // ELU
  }
  const size_t mbase = (size_t)(mt * 16 + half * 8);
  if (Outf) {
#pragma unroll
    for (int i = 0; i < 8; ++i) Outf[(mbase + i) * kD + col] = r[i];
  }
  if (Outb) {
#pragma unroll
    for (int i = 0; i < 8; ++i) Outb[(mbase + i) * kD + col] = (__bf16)r[i];
  }
}

template <int AMODE, int KVAL>
__global__ __launch_bounds__(128)
void wmma_gemm_elu(const __bf16* __restrict__ A0, const __bf16* __restrict__ A2,
                   const int* __restrict__ sidx, const int* __restrict__ didx,
                   const __bf16* __restrict__ Wp, const float* __restrict__ bias,
                   float* __restrict__ Outf, __bf16* __restrict__ Outb) {
  constexpr int kTiles = (KVAL + 31) / 32;
  const int lane = threadIdx.x & 31;
  const int wave = threadIdx.x >> 5;
  const int mt   = blockIdx.x;
  const int half = lane >> 4;
  const int lr   = lane & 15;
  const int row  = mt * 16 + lr;              // A-row owned by this lane
  const int nt0  = wave;                      // 4 waves x 2 N-tiles = 128 cols
  const int nt1  = wave + 4;

  int s0 = 0, s1 = 0;
  if constexpr (AMODE != 0) { s0 = sidx[row]; s1 = didx[row]; }

  v8f acc0 = {};
  v8f acc1 = {};
  const v16bf* WF = (const v16bf*)Wp;

#pragma unroll
  for (int kt = 0; kt < kTiles; ++kt) {
    // A fragment: 16-bit A 16x32 layout — lane holds row `row`,
    // runs k = kt*32 + half*8 + {0..7} and +16+{0..7}; each run = one 16B load.
    union { v16bf v; v8bf h[2]; } af;
    const int k0 = kt * 32 + half * 8;
    if constexpr (kTiles * 32 == KVAL || true) {
      if ((kt * 32 + 16) <= KVAL) {
        af.h[0] = *(const v8bf*)a_run_base<AMODE, KVAL>(A0, A2, s0, s1, row, k0);
      } else { v8bf z = {}; af.h[0] = z; }
      if ((kt * 32 + 32) <= KVAL) {
        af.h[1] = *(const v8bf*)a_run_base<AMODE, KVAL>(A0, A2, s0, s1, row, k0 + 16);
      } else { v8bf z = {}; af.h[1] = z; }
    }
    if (kt + 1 < kTiles) {    // prefetch next K-step's weight fragments
      __builtin_prefetch(&WF[((size_t)(kt + 1) * 8 + nt0) * 32 + lane], 0, 1);
    }
    const v16bf b0 = WF[((size_t)kt * 8 + nt0) * 32 + lane];
    acc0 = __builtin_amdgcn_wmma_f32_16x16x32_bf16(false, af.v, false, b0,
                                                   (short)0, acc0, false, false);
    const v16bf b1 = WF[((size_t)kt * 8 + nt1) * 32 + lane];
    acc1 = __builtin_amdgcn_wmma_f32_16x16x32_bf16(false, af.v, false, b1,
                                                   (short)0, acc1, false, false);
  }
  store_tile_elu(acc0, mt, nt0, lane, bias, Outf, Outb);
  store_tile_elu(acc1, mt, nt1, lane, bias, Outf, Outb);
}

// ---------------------- elementwise / graph kernels ------------------------
__global__ void fill_kernel(float* __restrict__ p, float v, size_t n) {
  size_t i = (size_t)blockIdx.x * blockDim.x + threadIdx.x;
  if (i < n) p[i] = v;
}

__global__ void f32_to_bf16(const float* __restrict__ in, __bf16* __restrict__ out,
                            size_t n) {
  size_t i = (size_t)blockIdx.x * blockDim.x + threadIdx.x;
  if (i < n) out[i] = (__bf16)in[i];
}

__global__ void count_deg(const int* __restrict__ dst, float* __restrict__ deg, int E) {
  int i = blockIdx.x * blockDim.x + threadIdx.x;
  if (i < E) atomicAdd(&deg[dst[i]], 1.0f);
}

__global__ void scatter_edges(const float* __restrict__ e, const int* __restrict__ dst,
                              float* __restrict__ agg, int E) {
  size_t t = (size_t)blockIdx.x * blockDim.x + threadIdx.x;
  if (t >= (size_t)E * kD) return;
  const int ed = (int)(t >> 7);
  const int d  = (int)(t & (kD - 1));
  atomicAdd(&agg[(size_t)dst[ed] * kD + d], e[t]);
}

// agg(fp32, atomic target) -> mean -> bf16 for the node GEMM
__global__ void div_deg_bf16(const float* __restrict__ agg, const float* __restrict__ deg,
                             __bf16* __restrict__ aggb, int N) {
  size_t t = (size_t)blockIdx.x * blockDim.x + threadIdx.x;
  if (t >= (size_t)N * kD) return;
  aggb[t] = (__bf16)(agg[t] / fmaxf(deg[t >> 7], 1.0f));
}

__global__ void copy_init(const float* __restrict__ e, float* __restrict__ x,
                          float* __restrict__ hh, __bf16* __restrict__ eb,
                          __bf16* __restrict__ xb, size_t n) {
  size_t i = (size_t)blockIdx.x * blockDim.x + threadIdx.x;
  if (i < n) {
    const float v = e[i];
    x[i] = v; hh[i] = v;
    const __bf16 vb = (__bf16)v;
    eb[i] = vb; xb[i] = vb;
  }
}

// y = x - (e - e_new); t = y - hh; x = hh + shrink(t); e = e_new
// also refreshes the bf16 shadows of e and x for the next GEMMs.
__global__ __launch_bounds__(256)
void prox_update(float* __restrict__ x, float* __restrict__ e,
                 const float* __restrict__ enew, const float* __restrict__ hh,
                 __bf16* __restrict__ eb, __bf16* __restrict__ xb,
                 int E, float lam) {
  const int row  = blockIdx.x * 8 + (threadIdx.x >> 5);
  const int lane = threadIdx.x & 31;
  if (row >= E) return;
  const size_t b = (size_t)row * kD;
  float t[4];
  float ss = 0.0f;
#pragma unroll
  for (int i = 0; i < 4; ++i) {
    const int d = lane + i * 32;
    const float en = enew[b + d];
    const float tv = (x[b + d] - (e[b + d] - en)) - hh[b + d];
    t[i] = tv;
    ss += tv * tv;
    e[b + d]  = en;                      // commit e = e_new (fp32 master)
    eb[b + d] = (__bf16)en;              // bf16 shadow for edge-MLP gather
  }
  for (int off = 16; off > 0; off >>= 1) ss += __shfl_xor(ss, off, 32);
  const float rn = sqrtf(ss);
  const float s  = (rn > 0.0f) ? fmaxf(rn - lam, 0.0f) / rn : 0.0f;
#pragma unroll
  for (int i = 0; i < 4; ++i) {
    const int d = lane + i * 32;
    const float xv = hh[b + d] + s * t[i];
    x[b + d]  = xv;
    xb[b + d] = (__bf16)xv;
  }
}

// final projection [E,128] @ [128,4] + b
__global__ void out_layer(const float* __restrict__ X, const float* __restrict__ W,
                          const float* __restrict__ b, float* __restrict__ out, int E) {
  int t = blockIdx.x * blockDim.x + threadIdx.x;
  if (t >= E * kClasses) return;
  const int m = t >> 2;
  const int c = t & 3;
  const float* xr = X + (size_t)m * kD;
  float s = b[c];
#pragma unroll 8
  for (int k = 0; k < kD; ++k) s += xr[k] * W[k * kClasses + c];
  out[t] = s;
}

// ---------------------------------------------------------------------------
extern "C" void kernel_launch(void* const* d_in, const int* in_sizes, int n_in,
                              void* d_out, int out_size, void* d_ws, size_t ws_size,
                              hipStream_t stream) {
  (void)in_sizes; (void)n_in; (void)out_size; (void)ws_size;
  const float* leaf = (const float*)d_in[0];
  const int*   src  = (const int*)d_in[1];
  const int*   dst  = (const int*)d_in[2];

  // Param flattening (setup_inputs insertion order, w then b, fc1 then fc2):
  // 3..6 init1, 7..10 init2, 11..14 pre, 15..18 node, 19..22 node_extra,
  // 23..26 edge, 27..30 final, 31..32 out.
  const int wIn[14] = {3, 5, 7, 9, 11, 13, 15, 17, 19, 21, 23, 25, 27, 29};
  const int bIn[14] = {4, 6, 8, 10, 12, 14, 16, 18, 20, 22, 24, 26, 28, 30};
  const int wK [14] = {kInFeat, kD, kD, kD, 2 * kD, kD, kD, kD, kD, kD, 3 * kD, kD, kD, kD};

  // -------- workspace carve: fp32 region, then bf16 region ----------------
  float* f = (float*)d_ws;
  size_t p = 0;
  float* deg  = f + p; p += kNodes;
  float* agg  = f + p; p += (size_t)kNodes * kD;   // atomic aggregation target
  float* e    = f + p; p += (size_t)kEdges * kD;   // edge state (fp32 master)
  float* enew = f + p; p += (size_t)kEdges * kD;
  float* x    = f + p; p += (size_t)kEdges * kD;
  float* hh   = f + p; p += (size_t)kEdges * kD;
  __bf16* bb = (__bf16*)(f + p);
  size_t q = 0;
  __bf16* leafb = bb + q; q += (size_t)kNodes * kInFeat;
  __bf16* hNb   = bb + q; q += (size_t)kNodes * kD;   // node embeddings h
  __bf16* tAb   = bb + q; q += (size_t)kNodes * kD;
  __bf16* tBb   = bb + q; q += (size_t)kNodes * kD;
  __bf16* aggb  = bb + q; q += (size_t)kNodes * kD;
  __bf16* ehidb = bb + q; q += (size_t)kEdges * kD;   // MLP hidden
  __bf16* eb    = bb + q; q += (size_t)kEdges * kD;   // bf16 shadow of e
  __bf16* xb    = bb + q; q += (size_t)kEdges * kD;   // bf16 shadow of x
  __bf16* Wp[14];
  for (int i = 0; i < 14; ++i) {
    Wp[i] = bb + q;
    q += (size_t)((wK[i] + 31) / 32) * (kD / 16) * 512;  // 32 lanes x 16 bf16
  }

  // -------- pack all weight matrices into WMMA fragment layout ------------
  for (int i = 0; i < 14; ++i) {
    const int frags = ((wK[i] + 31) / 32) * (kD / 16);
    pack_weight<<<frags, 32, 0, stream>>>((const float*)d_in[wIn[i]], Wp[i], wK[i], kD);
  }

  auto g16 = [&](const __bf16* A, int wi, __bf16* Ob) {
    wmma_gemm_elu<0, kInFeat><<<dim3(kNodes / 16), dim3(128), 0, stream>>>(
        A, nullptr, nullptr, nullptr, Wp[wi], (const float*)d_in[bIn[wi]], nullptr, Ob);
  };
  auto g128 = [&](const __bf16* A, int wi, float* Of, __bf16* Ob, int M) {
    wmma_gemm_elu<0, kD><<<dim3(M / 16), dim3(128), 0, stream>>>(
        A, nullptr, nullptr, nullptr, Wp[wi], (const float*)d_in[bIn[wi]], Of, Ob);
  };
  auto gcat2 = [&](const __bf16* A, int wi, __bf16* Ob) {
    wmma_gemm_elu<1, 2 * kD><<<dim3(kEdges / 16), dim3(128), 0, stream>>>(
        A, nullptr, src, dst, Wp[wi], (const float*)d_in[bIn[wi]], nullptr, Ob);
  };
  auto gcat3 = [&](const __bf16* A, const __bf16* E2, int wi, __bf16* Ob) {
    wmma_gemm_elu<2, 3 * kD><<<dim3(kEdges / 16), dim3(128), 0, stream>>>(
        A, E2, src, dst, Wp[wi], (const float*)d_in[bIn[wi]], nullptr, Ob);
  };

  // -------- degree (dst is static across iterations) ----------------------
  fill_kernel<<<(kNodes + 255) / 256, 256, 0, stream>>>(deg, 0.0f, (size_t)kNodes);
  count_deg<<<(kEdges + 255) / 256, 256, 0, stream>>>(dst, deg, kEdges);

  // -------- initial node MLP stack: h = init2(init1(leaf)) ----------------
  f32_to_bf16<<<(kNodes * kInFeat + 255) / 256, 256, 0, stream>>>(
      leaf, leafb, (size_t)kNodes * kInFeat);
  g16(leafb, 0, tAb);
  g128(tAb, 1, nullptr, tBb, kNodes);
  g128(tBb, 2, nullptr, tAb, kNodes);
  g128(tAb, 3, nullptr, hNb, kNodes);

  // -------- pre-blocks: e = mlp(pre, concat(h[src], h[dst])) --------------
  gcat2(hNb, 4, ehidb);
  g128(ehidb, 5, e, nullptr, kEdges);
  const size_t nEl = (size_t)kEdges * kD;
  copy_init<<<(int)((nEl + 255) / 256), 256, 0, stream>>>(e, x, hh, eb, xb, nEl);

  // -------- K proximal iterations -----------------------------------------
  const size_t nNd = (size_t)kNodes * kD;
  for (int it = 0; it < kIters; ++it) {
    // edge2node mean of current e
    fill_kernel<<<(int)((nNd + 255) / 256), 256, 0, stream>>>(agg, 0.0f, nNd);
    scatter_edges<<<(int)((nEl + 255) / 256), 256, 0, stream>>>(e, dst, agg, kEdges);
    div_deg_bf16<<<(int)((nNd + 255) / 256), 256, 0, stream>>>(agg, deg, aggb, kNodes);
    // node MLP + node_extra MLP
    g128(aggb, 6, nullptr, tAb, kNodes);
    g128(tAb,  7, nullptr, tBb, kNodes);
    g128(tBb,  8, nullptr, tAb, kNodes);
    g128(tAb,  9, nullptr, tBb, kNodes);
    // edge MLP on concat(node[src], node[dst], e) -> e_new
    gcat3(tBb, eb, 10, ehidb);
    g128(ehidb, 11, enew, nullptr, kEdges);
    // proximal L21 update of x; commits e = e_new + bf16 shadows
    prox_update<<<kEdges / 8, 256, 0, stream>>>(x, e, enew, hh, eb, xb, kEdges, kLam);
  }

  // -------- final MLP + output projection ---------------------------------
  g128(xb,    12, nullptr, ehidb, kEdges);
  g128(ehidb, 13, enew, nullptr, kEdges);
  out_layer<<<(kEdges * kClasses + 255) / 256, 256, 0, stream>>>(
      enew, (const float*)d_in[31], (const float*)d_in[32], (float*)d_out, kEdges);
}
